// Seq2SeqModel_9234179687333
// MI455X (gfx1250) — matrix operand order repaired and checked by reading, not compiled, not run on hardware
//
#include <hip/hip_runtime.h>
#include <hip/hip_bf16.h>
#include <cstdint>

// ---------------------------------------------------------------------------
// Seq2Seq 2-layer GRU (B=4096, R=1024, I=82->pad96, H=67) on gfx1250.
// Heavy GEMMs: v_wmma_f32_16x16x32_bf16, 256x128 block tile, 64x64 wave tile,
// double-buffered LDS, one barrier per K-chunk, and global_load_async_to_lds
// (ASYNCcnt-tracked) for zero-VGPR tile staging.
// ---------------------------------------------------------------------------

#define Bsz 4096
#define Rr 1024
#define Hh 67
#define Ii 82
#define Ipad 96
#define SRCm1 49
#define TGT 25

typedef __attribute__((ext_vector_type(16))) __bf16 v16bf;
typedef __attribute__((ext_vector_type(8)))  __bf16 v8bf;
typedef __attribute__((ext_vector_type(8)))  float  v8f;
typedef __attribute__((ext_vector_type(4)))  int    v4i;

union frag16 { v16bf v; v8bf h[2]; };

#if __has_builtin(__builtin_amdgcn_global_load_async_to_lds_b128) && \
    __has_builtin(__builtin_amdgcn_s_wait_asynccnt)
#define HAVE_ASYNC_LDS 1
// prototype (from clang diagnostic): (v4i AS1* src, v4i AS3* dst, Imm i32 off, Imm i32 cpol)
#define ASYNC_CP16(gsrc, ldst)                                          \
    __builtin_amdgcn_global_load_async_to_lds_b128(                     \
        (__attribute__((address_space(1))) v4i*)(gsrc),                 \
        (__attribute__((address_space(3))) v4i*)(ldst), 0, 0)
#else
#define HAVE_ASYNC_LDS 0
#endif

// ---------------- generic bf16 WMMA GEMM:  C[M,N] = A[M,K] * W[N,K]^T -------
// A row-major stride sA, W row-major stride sW, C row-major stride sC (f32).
// K multiple of 32; M multiple of 256; grid = (N/128, M/256).
#define BM 256
#define BN 128
#define KC 32
#define LDT 40   // padded LDS row stride (elements): 80B, staggers banks
#define ASZ (BM * LDT)
#define BSZ (BN * LDT)

__global__ __launch_bounds__(256, 1)
void gemm_bf16_wmma(const __bf16* __restrict__ A, const __bf16* __restrict__ W,
                    float* __restrict__ C, int sA, int sW, int sC, int K)
{
    __shared__ __bf16 As[2 * ASZ];
    __shared__ __bf16 Bs[2 * BSZ];

    const int tid  = threadIdx.x;
    const int wid  = tid >> 5;
    const int lane = tid & 31;
    const int g    = lane >> 4;     // lane group (0: lanes 0-15, 1: 16-31)
    const int ln   = lane & 15;

    const int m0 = blockIdx.y * BM;
    const int n0 = blockIdx.x * BN;

    const int wm = (wid >> 1) * 64; // wave tile: 64 rows
    const int wn = (wid & 1) * 64;  //            64 cols

    v8f acc[4][4] = {};

    // cooperative tile staging:
    //   A tile 256x32: thread t -> row t, 32 bf16 (4x 16B)
    //   B tile 128x32: 2 threads/row, 16 bf16 each (2x 16B)
    const int lrB = tid >> 1;
    const int lcB = (tid & 1) * 16;

    const __bf16* gA = A + (size_t)(m0 + tid) * sA;
    const __bf16* gW = W + (size_t)(n0 + lrB) * sW + lcB;

    // stage one 32-K chunk into LDS buffer (pA/pW already chunk-offset)
    auto loadTiles = [&](const __bf16* pA, const __bf16* pW, __bf16* dA, __bf16* dB) {
#if HAVE_ASYNC_LDS
        ASYNC_CP16(pA,      &dA[tid * LDT]);
        ASYNC_CP16(pA + 8,  &dA[tid * LDT + 8]);
        ASYNC_CP16(pA + 16, &dA[tid * LDT + 16]);
        ASYNC_CP16(pA + 24, &dA[tid * LDT + 24]);
        ASYNC_CP16(pW,      &dB[lrB * LDT + lcB]);
        ASYNC_CP16(pW + 8,  &dB[lrB * LDT + lcB + 8]);
#else
        uint4 a0 = *reinterpret_cast<const uint4*>(pA);
        uint4 a1 = *reinterpret_cast<const uint4*>(pA + 8);
        uint4 a2 = *reinterpret_cast<const uint4*>(pA + 16);
        uint4 a3 = *reinterpret_cast<const uint4*>(pA + 24);
        uint4 w0 = *reinterpret_cast<const uint4*>(pW);
        uint4 w1 = *reinterpret_cast<const uint4*>(pW + 8);
        *reinterpret_cast<uint4*>(&dA[tid * LDT])      = a0;
        *reinterpret_cast<uint4*>(&dA[tid * LDT + 8])  = a1;
        *reinterpret_cast<uint4*>(&dA[tid * LDT + 16]) = a2;
        *reinterpret_cast<uint4*>(&dA[tid * LDT + 24]) = a3;
        *reinterpret_cast<uint4*>(&dB[lrB * LDT + lcB])     = w0;
        *reinterpret_cast<uint4*>(&dB[lrB * LDT + lcB + 8]) = w1;
#endif
    };

    auto syncTiles = [&]() {
#if HAVE_ASYNC_LDS
        __builtin_amdgcn_s_wait_asynccnt(0);  // our async LDS writes landed
#endif
        __syncthreads();                      // everyone's writes visible
    };

    // 16 WMMAs on one staged chunk
    auto computeChunk = [&](const __bf16* Ab, const __bf16* Bb) {
        frag16 a[4], b[4];
        #pragma unroll
        for (int i = 0; i < 4; ++i) {
            // A frag (ISA 16-bit A 16x32 layout): lane row = ln, hi half at K+16
            const __bf16* ap = &Ab[(wm + i * 16 + ln) * LDT + g * 8];
            a[i].h[0] = *reinterpret_cast<const v8bf*>(ap);
            a[i].h[1] = *reinterpret_cast<const v8bf*>(ap + 16);
        }
        #pragma unroll
        for (int j = 0; j < 4; ++j) {
            // B frag: column n = ln, K layout matches A (g*8, hi half at K+16)
            const __bf16* bp = &Bb[(wn + j * 16 + ln) * LDT + g * 8];
            b[j].h[0] = *reinterpret_cast<const v8bf*>(bp);
            b[j].h[1] = *reinterpret_cast<const v8bf*>(bp + 16);
        }
        #pragma unroll
        for (int i = 0; i < 4; ++i)
            #pragma unroll
            for (int j = 0; j < 4; ++j)
                acc[i][j] = __builtin_amdgcn_wmma_f32_16x16x32_bf16(
                    false, a[i].v, false, b[j].v, (short)0, acc[i][j], false, false);
    };

    // ---- software pipeline: nk-1 prefetching iterations + peeled tail ----
    loadTiles(gA, gW, As, Bs);   // chunk 0 -> buffer 0
    syncTiles();

    int buf = 0;
    const int nk = K / KC;
    for (int ck = 0; ck < nk - 1; ++ck) {
        const int koff = (ck + 1) * KC;
        loadTiles(gA + koff, gW + koff, &As[(buf ^ 1) * ASZ], &Bs[(buf ^ 1) * BSZ]);
        computeChunk(&As[buf * ASZ], &Bs[buf * BSZ]);
        syncTiles();             // prefetch landed; all reads of buf done
        buf ^= 1;
    }
    computeChunk(&As[buf * ASZ], &Bs[buf * BSZ]);

    // C layout: VGPR v -> row v + 8*g, col = ln
    #pragma unroll
    for (int i = 0; i < 4; ++i)
        #pragma unroll
        for (int j = 0; j < 4; ++j) {
            float* cp = C + (size_t)(m0 + wm + i * 16 + g * 8) * sC
                          + (n0 + wn + j * 16 + ln);
            #pragma unroll
            for (int v = 0; v < 8; ++v)
                cp[(size_t)v * sC] = acc[i][j][v];
        }
}

// ---------------- weight convert fp32 -> bf16 with row/col zero padding ----
__global__ void cvt_pad_kernel(const float* __restrict__ src, __bf16* __restrict__ dst,
                               int rows, int srcK, int dstK, int srcRows)
{
    int idx = blockIdx.x * blockDim.x + threadIdx.x;
    if (idx >= rows * dstK) return;
    int r = idx / dstK, k = idx - r * dstK;
    float v = (r < srcRows && k < srcK) ? src[(size_t)r * srcK + k] : 0.0f;
    dst[idx] = (__bf16)v;
}

// ---------------- zero hidden state ----------------------------------------
__global__ void zero_state_kernel(float* h1, float* h2, __bf16* h1b, __bf16* h2b)
{
    int idx = blockIdx.x * blockDim.x + threadIdx.x;
    if (idx >= Bsz * Rr) return;
    h1[idx] = 0.0f; h2[idx] = 0.0f;
    h1b[idx] = (__bf16)0.0f; h2b[idx] = (__bf16)0.0f;
}

// ---------------- encoder input pack (pad 82 -> 96, bf16) ------------------
__global__ void pack_x_enc_kernel(const float* __restrict__ enc, __bf16* __restrict__ xb, int t)
{
    int idx = blockIdx.x * blockDim.x + threadIdx.x;
    if (idx >= Bsz * Ipad) return;
    int b = idx / Ipad, k = idx - b * Ipad;
    float v = (k < Ii) ? enc[((size_t)b * SRCm1 + t) * Ii + k] : 0.0f;
    xb[idx] = (__bf16)v;
}

// ---------------- decoder input pack: concat(prev_out[:H], dec_t[H:]) ------
__global__ void pack_x_dec_kernel(const float* __restrict__ dec, const float* __restrict__ out,
                                  __bf16* __restrict__ xb, int t)
{
    int idx = blockIdx.x * blockDim.x + threadIdx.x;
    if (idx >= Bsz * Ipad) return;
    int b = idx / Ipad, k = idx - b * Ipad;
    float v = 0.0f;
    if (k < Hh)
        v = (t == 0) ? dec[(size_t)b * TGT * Ii + k]
                     : out[((size_t)b * TGT + (t - 1)) * Hh + k];
    else if (k < Ii)
        v = dec[((size_t)b * TGT + t) * Ii + k];
    xb[idx] = (__bf16)v;
}

// ---------------- GRU gates (torch GRUCell, gate order r,z,n) --------------
__global__ void gru_gate_kernel(const float* __restrict__ gi, const float* __restrict__ gh,
                                const float* __restrict__ bi, const float* __restrict__ bh,
                                float* __restrict__ h, __bf16* __restrict__ hb)
{
    int idx = blockIdx.x * blockDim.x + threadIdx.x;
    if (idx >= Bsz * Rr) return;
    int b = idx >> 10, j = idx & (Rr - 1);
    size_t base = (size_t)b * 3 * Rr;
    float ir  = gi[base + j]            + bi[j];
    float iz  = gi[base + Rr + j]       + bi[Rr + j];
    float inn = gi[base + 2 * Rr + j]   + bi[2 * Rr + j];
    float hr  = gh[base + j]            + bh[j];
    float hz  = gh[base + Rr + j]       + bh[Rr + j];
    float hn  = gh[base + 2 * Rr + j]   + bh[2 * Rr + j];
    float r = 1.0f / (1.0f + __expf(-(ir + hr)));
    float z = 1.0f / (1.0f + __expf(-(iz + hz)));
    float n = tanhf(inn + r * hn);
    float ho = h[idx];
    float hnew = (1.0f - z) * n + z * ho;
    h[idx]  = hnew;
    hb[idx] = (__bf16)hnew;
}

// ---------------- concat [h1 | h2] (bf16) for the FC GEMM ------------------
__global__ void pack_state_kernel(const __bf16* __restrict__ h1b, const __bf16* __restrict__ h2b,
                                  __bf16* __restrict__ stateb)
{
    int idx = blockIdx.x * blockDim.x + threadIdx.x;
    if (idx >= Bsz * 2 * Rr) return;
    int b = idx >> 11, k = idx & (2 * Rr - 1);
    stateb[idx] = (k < Rr) ? h1b[(size_t)b * Rr + k] : h2b[(size_t)b * Rr + (k - Rr)];
}

// ---------------- out[b,t,:] = prev[:H] + fc + bfc -------------------------
__global__ void fc_finalize_kernel(const float* __restrict__ fcbuf, const float* __restrict__ bfc,
                                   const float* __restrict__ dec, float* __restrict__ out, int t)
{
    int idx = blockIdx.x * blockDim.x + threadIdx.x;
    if (idx >= Bsz * Hh) return;
    int b = idx / Hh, hh = idx - b * Hh;
    float prev = (t == 0) ? dec[(size_t)b * TGT * Ii + hh]
                          : out[((size_t)b * TGT + (t - 1)) * Hh + hh];
    out[((size_t)b * TGT + t) * Hh + hh] = prev + fcbuf[(size_t)b * 128 + hh] + bfc[hh];
}

// ---------------------------------------------------------------------------
static inline int cdiv(int a, int b) { return (a + b - 1) / b; }

extern "C" void kernel_launch(void* const* d_in, const int* in_sizes, int n_in,
                              void* d_out, int out_size, void* d_ws, size_t ws_size,
                              hipStream_t stream)
{
    const float* enc = (const float*)d_in[0];
    const float* dec = (const float*)d_in[1];
    const float* Wi1 = (const float*)d_in[2];
    const float* Wh1 = (const float*)d_in[3];
    const float* bi1 = (const float*)d_in[4];
    const float* bh1 = (const float*)d_in[5];
    const float* Wi2 = (const float*)d_in[6];
    const float* Wh2 = (const float*)d_in[7];
    const float* bi2 = (const float*)d_in[8];
    const float* bh2 = (const float*)d_in[9];
    const float* Wfc = (const float*)d_in[10];
    const float* bfc = (const float*)d_in[11];
    float* out = (float*)d_out;

    // -------- workspace carve-up (~190 MB), 256B aligned ------------------
    char* ws = (char*)d_ws;
    size_t off = 0;
    auto take = [&](size_t bytes) -> char* {
        char* p = ws + off;
        off = (off + bytes + 255) & ~(size_t)255;
        return p;
    };
    __bf16* wi1b   = (__bf16*)take((size_t)3 * Rr * Ipad * 2);
    __bf16* wh1b   = (__bf16*)take((size_t)3 * Rr * Rr * 2);
    __bf16* wi2b   = (__bf16*)take((size_t)3 * Rr * Rr * 2);
    __bf16* wh2b   = (__bf16*)take((size_t)3 * Rr * Rr * 2);
    __bf16* wfcb   = (__bf16*)take((size_t)128 * 2 * Rr * 2);
    __bf16* xb     = (__bf16*)take((size_t)Bsz * Ipad * 2);
    float*  h1     = (float*) take((size_t)Bsz * Rr * 4);
    float*  h2     = (float*) take((size_t)Bsz * Rr * 4);
    __bf16* h1b    = (__bf16*)take((size_t)Bsz * Rr * 2);
    __bf16* h2b    = (__bf16*)take((size_t)Bsz * Rr * 2);
    __bf16* stateb = (__bf16*)take((size_t)Bsz * 2 * Rr * 2);
    float*  gi     = (float*) take((size_t)Bsz * 3 * Rr * 4);
    float*  gh     = (float*) take((size_t)Bsz * 3 * Rr * 4);
    float*  fcbuf  = (float*) take((size_t)Bsz * 128 * 4);
    (void)ws_size; (void)in_sizes; (void)n_in; (void)out_size;

    const int TB = 256;

    // -------- one-time (per launch) weight conversion to bf16 -------------
    cvt_pad_kernel<<<cdiv(3 * Rr * Ipad, TB), TB, 0, stream>>>(Wi1, wi1b, 3 * Rr, Ii, Ipad, 3 * Rr);
    cvt_pad_kernel<<<cdiv(3 * Rr * Rr, TB), TB, 0, stream>>>(Wh1, wh1b, 3 * Rr, Rr, Rr, 3 * Rr);
    cvt_pad_kernel<<<cdiv(3 * Rr * Rr, TB), TB, 0, stream>>>(Wi2, wi2b, 3 * Rr, Rr, Rr, 3 * Rr);
    cvt_pad_kernel<<<cdiv(3 * Rr * Rr, TB), TB, 0, stream>>>(Wh2, wh2b, 3 * Rr, Rr, Rr, 3 * Rr);
    cvt_pad_kernel<<<cdiv(128 * 2 * Rr, TB), TB, 0, stream>>>(Wfc, wfcb, 128, 2 * Rr, 2 * Rr, Hh);
    zero_state_kernel<<<cdiv(Bsz * Rr, TB), TB, 0, stream>>>(h1, h2, h1b, h2b);

    dim3 ggrid(3 * Rr / BN, Bsz / BM);   // 24 x 16 workgroups per big GEMM
    dim3 gblock(256);
    dim3 fcgrid(1, Bsz / BM);            // N=128 tile for the FC GEMM

    // -------------------------- encoder -----------------------------------
    for (int t = 0; t < SRCm1; ++t) {
        pack_x_enc_kernel<<<cdiv(Bsz * Ipad, TB), TB, 0, stream>>>(enc, xb, t);
        gemm_bf16_wmma<<<ggrid, gblock, 0, stream>>>(xb,  wi1b, gi, Ipad, Ipad, 3 * Rr, Ipad);
        gemm_bf16_wmma<<<ggrid, gblock, 0, stream>>>(h1b, wh1b, gh, Rr,   Rr,   3 * Rr, Rr);
        gru_gate_kernel<<<cdiv(Bsz * Rr, TB), TB, 0, stream>>>(gi, gh, bi1, bh1, h1, h1b);
        gemm_bf16_wmma<<<ggrid, gblock, 0, stream>>>(h1b, wi2b, gi, Rr,   Rr,   3 * Rr, Rr);
        gemm_bf16_wmma<<<ggrid, gblock, 0, stream>>>(h2b, wh2b, gh, Rr,   Rr,   3 * Rr, Rr);
        gru_gate_kernel<<<cdiv(Bsz * Rr, TB), TB, 0, stream>>>(gi, gh, bi2, bh2, h2, h2b);
    }

    // -------------------------- decoder -----------------------------------
    for (int t = 0; t < TGT; ++t) {
        pack_x_dec_kernel<<<cdiv(Bsz * Ipad, TB), TB, 0, stream>>>(dec, out, xb, t);
        gemm_bf16_wmma<<<ggrid, gblock, 0, stream>>>(xb,  wi1b, gi, Ipad, Ipad, 3 * Rr, Ipad);
        gemm_bf16_wmma<<<ggrid, gblock, 0, stream>>>(h1b, wh1b, gh, Rr,   Rr,   3 * Rr, Rr);
        gru_gate_kernel<<<cdiv(Bsz * Rr, TB), TB, 0, stream>>>(gi, gh, bi1, bh1, h1, h1b);
        gemm_bf16_wmma<<<ggrid, gblock, 0, stream>>>(h1b, wi2b, gi, Rr,   Rr,   3 * Rr, Rr);
        gemm_bf16_wmma<<<ggrid, gblock, 0, stream>>>(h2b, wh2b, gh, Rr,   Rr,   3 * Rr, Rr);
        gru_gate_kernel<<<cdiv(Bsz * Rr, TB), TB, 0, stream>>>(gi, gh, bi2, bh2, h2, h2b);

        pack_state_kernel<<<cdiv(Bsz * 2 * Rr, TB), TB, 0, stream>>>(h1b, h2b, stateb);
        gemm_bf16_wmma<<<fcgrid, gblock, 0, stream>>>(stateb, wfcb, fcbuf, 2 * Rr, 2 * Rr, 128, 2 * Rr);
        fc_finalize_kernel<<<cdiv(Bsz * Hh, TB), TB, 0, stream>>>(fcbuf, bfc, dec, out, t);
    }
}
